// GATNodeEdgePrediction_4243427688595
// MI455X (gfx1250) — matrix operand order repaired and checked by reading, not compiled, not run on hardware
//
#include <hip/hip_runtime.h>
#include <hip/hip_bf16.h>
#include <math.h>

#define N_NODES 50000
#define N_EDGES 800000

typedef __attribute__((ext_vector_type(16))) __bf16 v16bf;
typedef __attribute__((ext_vector_type(8)))  float  v8f;
typedef __attribute__((ext_vector_type(4)))  unsigned u32x4;
typedef __attribute__((ext_vector_type(8)))  int      i32x8;
typedef __attribute__((ext_vector_type(4)))  int      i32x4;

#define CDIV(a, b) (((a) + (b) - 1) / (b))

// 32-byte POD for bit-casting two b128 LDS loads into a v16bf fragment
struct U8x32 { uint4 a, b; };

// Packed-weight tile geometry: 64 columns x 20 dwords (16 k-pairs + 4 pad).
// Stride 20 dwords -> 16B-aligned rows + conflict-free LDS bank rotation.
#define BSTRIDE 20
#define TILE_DW (64 * BSTRIDE)   // 1280 dwords = 5120 bytes per (32k x 64n) tile

#if defined(__gfx1250__) && __has_builtin(__builtin_amdgcn_tensor_load_to_lds)
#define HAVE_TDM 1
#else
#define HAVE_TDM 0
#endif

// ---------------------------------------------------------------------------
// Weight pre-pack: W[K,N] f32 -> PW tiles. Tile ti = nb*(K/32)+kb, 1280 dwords.
// Within tile: dword [nn*20 + kp] = { bf16(W[kb*32+2kp][nb*64+nn]),
//                                     bf16(W[kb*32+2kp+1][nb*64+nn]) }, kp<16.
// This is exactly the LDS image the GEMM fragments read -> staging is a
// contiguous copy (TDM-friendly).
// ---------------------------------------------------------------------------
__global__ void k_packW(const float* __restrict__ W, unsigned* __restrict__ PW,
                        int K, int N) {
  int total = (K / 32) * (N / 64) * TILE_DW;
  int t = blockIdx.x * blockDim.x + threadIdx.x;
  if (t >= total) return;
  int ti = t / TILE_DW, r = t % TILE_DW;
  int nn = r / BSTRIDE, kp = r % BSTRIDE;
  int kb = ti % (K / 32), nb = ti / (K / 32);
  unsigned val = 0u;
  if (kp < 16) {
    int k = kb * 32 + 2 * kp;
    int n = nb * 64 + nn;
    union { unsigned u; __bf16 h[2]; } pk;
    pk.h[0] = (__bf16)W[(size_t)k * N + n];
    pk.h[1] = (__bf16)W[(size_t)(k + 1) * N + n];
    val = pk.u;
  }
  PW[t] = val;
}

// ---------------------------------------------------------------------------
// Generic GEMM: C[M,N] = A[M,K] @ B[K,N] (+bias), B given pre-packed (PW).
// mode 0: C = acc (+bias)
// mode 1: C = gate * sigmoid(acc + bias)   (requires N == K, gate == A input)
// mode 2: C = relu(acc + bias)
// Requirements: K % 32 == 0, N % 64 == 0. M guarded.
// Block: 256 threads (8 waves). Block tile: 128 rows x 64 cols.
// Per K-step: TDM copies one 5KB packed tile into LDS; each lane's B fragment
// is two ds_load_b128; A fragment is four global_load_b128 + cvt_pk.
// ---------------------------------------------------------------------------
__global__ void gemm_bf16_wmma(const float* __restrict__ A,
                               const unsigned* __restrict__ PW,
                               const float* __restrict__ bias,
                               const float* __restrict__ gate,
                               float* __restrict__ C,
                               int M, int N, int K, int mode) {
  __shared__ unsigned sB[TILE_DW];

  const int tid  = threadIdx.x;
  const int w    = tid >> 5;
  const int lane = tid & 31;
  const int h    = lane >> 4;   // lane half: K-subset (A/B), M-half (D)
  const int ml   = lane & 15;   // A row within tile / D column within tile

  const int tileM = blockIdx.x * 128;
  const int nb    = blockIdx.y;
  const int n0    = nb * 64;
  const int row   = tileM + 16 * w + ml;
  const int KB    = K / 32;

  v8f acc[4];
#pragma unroll
  for (int t = 0; t < 4; ++t)
#pragma unroll
    for (int i = 0; i < 8; ++i) acc[t][i] = 0.0f;

  for (int kb = 0; kb < KB; ++kb) {
    const int k0 = kb * 32;
    const size_t tileBase = ((size_t)nb * KB + kb) * TILE_DW;
    __syncthreads();   // previous tile fully consumed
#if HAVE_TDM
    if (w == 0) {
      // 1-D tensor DMA: copy TILE_DW dwords global -> LDS.
      unsigned ldsoff = (unsigned)(unsigned long long)(void*)&sB[0];
      unsigned long long ga = (unsigned long long)(const void*)(PW + tileBase);
      u32x4 g0;
      g0[0] = 1u;                                         // count=1, user desc
      g0[1] = ldsoff;                                     // lds_addr
      g0[2] = (unsigned)ga;                               // global_addr[31:0]
      g0[3] = (unsigned)((ga >> 32) & 0x01FFFFFFull) | 0x80000000u; // [56:32]|type=2
      i32x8 g1;
      g1[0] = 0x00020000;                                 // data_size=2 (4B)
      g1[1] = (int)((TILE_DW & 0xFFFFu) << 16);           // tensor_dim0 lo16
      g1[2] = (int)((TILE_DW >> 16) | (1u << 16));        // td0 hi16 | tensor_dim1=1
      g1[3] = (int)((TILE_DW & 0xFFFFu) << 16);           // tile_dim0 = TILE_DW
      g1[4] = 1;                                          // tile_dim1=1, tile_dim2=0
      g1[5] = (int)TILE_DW;                               // dim0_stride lo32
      g1[6] = 0;
      g1[7] = 0;
      i32x4 gz;
      gz[0] = 0; gz[1] = 0; gz[2] = 0; gz[3] = 0;
      i32x8 gz8;
      gz8[0] = 0; gz8[1] = 0; gz8[2] = 0; gz8[3] = 0;
      gz8[4] = 0; gz8[5] = 0; gz8[6] = 0; gz8[7] = 0;
      __builtin_amdgcn_tensor_load_to_lds(g0, g1, gz, gz, gz8, 0);
      __builtin_amdgcn_s_wait_tensorcnt(0);
    }
#else
#pragma unroll
    for (int r = 0; r < 5; ++r) {
      int idx = tid + 256 * r;          // 0..1279
      sB[idx] = PW[tileBase + idx];
    }
#endif
    __syncthreads();

    // A fragment: lane holds row `row`, K offsets {8h..8h+7} U {16+8h..16+8h+7}
    v16bf a;
    if (row < M) {
      const float* ap = A + (size_t)row * K + k0;
      float4 f0 = *(const float4*)(ap + 8 * h);
      float4 f1 = *(const float4*)(ap + 8 * h + 4);
      float4 f2 = *(const float4*)(ap + 16 + 8 * h);
      float4 f3 = *(const float4*)(ap + 16 + 8 * h + 4);
      a[0]  = (__bf16)f0.x; a[1]  = (__bf16)f0.y; a[2]  = (__bf16)f0.z; a[3]  = (__bf16)f0.w;
      a[4]  = (__bf16)f1.x; a[5]  = (__bf16)f1.y; a[6]  = (__bf16)f1.z; a[7]  = (__bf16)f1.w;
      a[8]  = (__bf16)f2.x; a[9]  = (__bf16)f2.y; a[10] = (__bf16)f2.z; a[11] = (__bf16)f2.w;
      a[12] = (__bf16)f3.x; a[13] = (__bf16)f3.y; a[14] = (__bf16)f3.z; a[15] = (__bf16)f3.w;
    } else {
#pragma unroll
      for (int i = 0; i < 16; ++i) a[i] = (__bf16)0.0f;
    }

#pragma unroll
    for (int t = 0; t < 4; ++t) {
      // B fragment for col tile t: VGPR j <- sB[(16t+ml)*BSTRIDE + 4h + j] (j<4), + 8 + 4h + (j-4) (j>=4)
      int base = (16 * t + ml) * BSTRIDE + 4 * h;
      U8x32 raw;
      raw.a = *(const uint4*)&sB[base];
      raw.b = *(const uint4*)&sB[base + 8];
      v16bf b = __builtin_bit_cast(v16bf, raw);
      acc[t] = __builtin_amdgcn_wmma_f32_16x16x32_bf16(
          false, a, false, b, (short)0, acc[t], false, false);
    }
  }

  // Epilogue. D layout: VGPR r -> row tileM+16w + r + 8h, col = n0 + 16t + ml.
#pragma unroll
  for (int t = 0; t < 4; ++t) {
    int col = n0 + 16 * t + ml;
#pragma unroll
    for (int r = 0; r < 8; ++r) {
      int rr = tileM + 16 * w + r + 8 * h;
      if (rr < M) {
        float v = acc[t][r];
        if (bias) v += bias[col];
        if (mode == 1) {
          float g = gate[(size_t)rr * N + col];
          v = g * (1.0f / (1.0f + __expf(-v)));
        } else if (mode == 2) {
          v = fmaxf(v, 0.0f);
        }
        C[(size_t)rr * N + col] = v;
      }
    }
  }
}

// ---------------------------------------------------------------------------
// Order-preserving float <-> uint key for atomic max
// ---------------------------------------------------------------------------
__device__ __forceinline__ unsigned fkey(float f) {
  unsigned u = __float_as_uint(f);
  return (u & 0x80000000u) ? ~u : (u | 0x80000000u);
}
__device__ __forceinline__ float funkey(unsigned u) {
  unsigned v = (u & 0x80000000u) ? (u & 0x7FFFFFFFu) : ~u;
  return __uint_as_float(v);
}

// degree + edge-attr sum per destination
__global__ void k_deg(const int* __restrict__ src, const int* __restrict__ dst,
                      const float* __restrict__ ea, float* __restrict__ deg,
                      float* __restrict__ eas, int E) {
  int e = blockIdx.x * blockDim.x + threadIdx.x;
  if (e >= E) return;
  int d = dst[e];
  atomicAdd(&deg[d], 1.0f);
  atomicAdd(&eas[d], ea[e]);
}

__global__ void k_loopattr(const float* __restrict__ deg, const float* __restrict__ eas,
                           float* __restrict__ loopa, int n) {
  int i = blockIdx.x * blockDim.x + threadIdx.x;
  if (i >= n) return;
  loopa[i] = eas[i] / fmaxf(deg[i], 1.0f);
}

// per-head edge coefficient: ch[h] = sum_c We[h*C+c] * ae[h*C+c]
__global__ void k_ch(const float* __restrict__ We, const float* __restrict__ ae,
                     float* __restrict__ ch, int H, int C) {
  int h = threadIdx.x;
  if (h >= H) return;
  float s = 0.0f;
  for (int c = 0; c < C; ++c) s += We[h * C + c] * ae[h * C + c];
  ch[h] = s;
}

// a_src[n,h], a_dst[n,h]: per-head dot over C
__global__ void k_att(const float* __restrict__ Hb, const float* __restrict__ as_,
                      const float* __restrict__ ad_, float* __restrict__ asrc,
                      float* __restrict__ adst, int n, int H, int C) {
  int t = blockIdx.x * blockDim.x + threadIdx.x;
  if (t >= n * H) return;
  int node = t / H, h = t % H;
  const float* hp = Hb + (size_t)node * H * C + h * C;
  float s = 0.0f, d = 0.0f;
  for (int c = 0; c < C; ++c) {
    float hv = hp[c];
    s += hv * as_[h * C + c];
    d += hv * ad_[h * C + c];
  }
  asrc[t] = s;
  adst[t] = d;
}

// edge logits + atomic segment max
__global__ void k_lg(const int* __restrict__ src, const int* __restrict__ dst,
                     const float* __restrict__ ea, const float* __restrict__ asrc,
                     const float* __restrict__ adst, const float* __restrict__ ch,
                     float* __restrict__ LG, unsigned* __restrict__ MKEY, int E, int H) {
  int t = blockIdx.x * blockDim.x + threadIdx.x;
  if (t >= E * H) return;
  int e = t / H, h = t % H;
  int s = src[e], d = dst[e];
  float v = asrc[s * H + h] + adst[d * H + h] + ea[e] * ch[h];
  v = (v > 0.0f) ? v : 0.2f * v;
  LG[t] = v;
  atomicMax(&MKEY[d * H + h], fkey(v));
}

// self-loop logits + atomic segment max
__global__ void k_lg_self(const float* __restrict__ asrc, const float* __restrict__ adst,
                          const float* __restrict__ loopa, const float* __restrict__ ch,
                          float* __restrict__ LGS, unsigned* __restrict__ MKEY, int n, int H) {
  int t = blockIdx.x * blockDim.x + threadIdx.x;
  if (t >= n * H) return;
  int node = t / H, h = t % H;
  float v = asrc[t] + adst[t] + loopa[node] * ch[h];
  v = (v > 0.0f) ? v : 0.2f * v;
  LGS[t] = v;
  atomicMax(&MKEY[t], fkey(v));
}

// p = exp(lg - m); store p in-place; z += p
__global__ void k_p(const int* __restrict__ dst, float* __restrict__ LG,
                    const unsigned* __restrict__ MKEY, float* __restrict__ Z, int E, int H) {
  int t = blockIdx.x * blockDim.x + threadIdx.x;
  if (t >= E * H) return;
  int e = t / H, h = t % H;
  int d = dst[e];
  float m = funkey(MKEY[d * H + h]);
  float p = __expf(LG[t] - m);
  LG[t] = p;
  atomicAdd(&Z[d * H + h], p);
}

__global__ void k_p_self(float* __restrict__ LGS, const unsigned* __restrict__ MKEY,
                         float* __restrict__ Z, int n, int H) {
  int t = blockIdx.x * blockDim.x + threadIdx.x;
  if (t >= n * H) return;
  float m = funkey(MKEY[t]);
  float p = __expf(LGS[t] - m);
  LGS[t] = p;
  atomicAdd(&Z[t], p);
}

// out[dst] += (p/z) * h[src], float4 per thread, atomic
__global__ void k_agg(const int* __restrict__ src, const int* __restrict__ dst,
                      const float* __restrict__ LG, const float* __restrict__ Z,
                      const float* __restrict__ Hb, float* __restrict__ OUT,
                      int E, int H, int C) {
  int HC4 = (H * C) >> 2;
  long t = (long)blockIdx.x * blockDim.x + threadIdx.x;
  if (t >= (long)E * HC4) return;
  int e = (int)(t / HC4), j = (int)(t % HC4);
  int c4 = j * 4;
  int h = c4 / C;
  int s = src[e], d = dst[e];
  float alpha = LG[e * H + h] / Z[d * H + h];
  float4 hv = *(const float4*)(Hb + (size_t)s * H * C + c4);
  float* op = OUT + (size_t)d * H * C + c4;
  atomicAdd(op + 0, alpha * hv.x);
  atomicAdd(op + 1, alpha * hv.y);
  atomicAdd(op + 2, alpha * hv.z);
  atomicAdd(op + 3, alpha * hv.w);
}

// self-loop aggregation (one thread per slot; kernels are serialized so plain add)
__global__ void k_agg_self(const float* __restrict__ LGS, const float* __restrict__ Z,
                           const float* __restrict__ Hb, float* __restrict__ OUT,
                           int n, int H, int C) {
  int HC4 = (H * C) >> 2;
  long t = (long)blockIdx.x * blockDim.x + threadIdx.x;
  if (t >= (long)n * HC4) return;
  int v = (int)(t / HC4), j = (int)(t % HC4);
  int c4 = j * 4;
  int h = c4 / C;
  float alpha = LGS[v * H + h] / Z[v * H + h];
  float4 hv = *(const float4*)(Hb + (size_t)v * H * C + c4);
  float* op = OUT + (size_t)v * H * C + c4;
  op[0] += alpha * hv.x;
  op[1] += alpha * hv.y;
  op[2] += alpha * hv.z;
  op[3] += alpha * hv.w;
}

// out = leaky_relu(acc + bias, 0.01) + res
__global__ void k_combine(const float* __restrict__ acc, const float* __restrict__ bias,
                          const float* __restrict__ res, float* __restrict__ out,
                          int n, int W) {
  int t = blockIdx.x * blockDim.x + threadIdx.x;
  if (t >= n * W) return;
  float v = acc[t] + bias[t % W];
  v = (v > 0.0f) ? v : 0.01f * v;
  out[t] = v + res[t];
}

// ---------------------------------------------------------------------------
// Host side
// ---------------------------------------------------------------------------
static size_t packed_dwords(int K, int N) {
  return (size_t)(K / 32) * (N / 64) * TILE_DW;
}

static void pack_weight(const float* W, unsigned* PW, int K, int N, hipStream_t stream) {
  int total = (int)packed_dwords(K, N);
  k_packW<<<CDIV(total, 256), 256, 0, stream>>>(W, PW, K, N);
}

static void gat_layer(const float* X, int K, const unsigned* PW, const float* We,
                      const float* as_, const float* ad_, const float* ae_,
                      int H, int C,
                      const int* src, const int* dst, const float* ea,
                      const float* LOOPA,
                      float* Hb, float* ASRC, float* ADST, unsigned* MKEY,
                      float* ZSUM, float* LG, float* LGS, float* CH, float* OUT,
                      hipStream_t stream) {
  const int n = N_NODES, E = N_EDGES;
  const int HC = H * C;
  // h = X @ W
  gemm_bf16_wmma<<<dim3(CDIV(n, 128), HC / 64), 256, 0, stream>>>(
      X, PW, nullptr, nullptr, Hb, n, HC, K, 0);
  k_ch<<<1, 32, 0, stream>>>(We, ae_, CH, H, C);
  k_att<<<CDIV(n * H, 256), 256, 0, stream>>>(Hb, as_, ad_, ASRC, ADST, n, H, C);
  (void)hipMemsetAsync(MKEY, 0, (size_t)n * H * sizeof(unsigned), stream);
  (void)hipMemsetAsync(ZSUM, 0, (size_t)n * H * sizeof(float), stream);
  (void)hipMemsetAsync(OUT, 0, (size_t)n * HC * sizeof(float), stream);
  k_lg<<<CDIV(E * H, 256), 256, 0, stream>>>(src, dst, ea, ASRC, ADST, CH, LG, MKEY, E, H);
  k_lg_self<<<CDIV(n * H, 256), 256, 0, stream>>>(ASRC, ADST, LOOPA, CH, LGS, MKEY, n, H);
  k_p<<<CDIV(E * H, 256), 256, 0, stream>>>(dst, LG, MKEY, ZSUM, E, H);
  k_p_self<<<CDIV(n * H, 256), 256, 0, stream>>>(LGS, MKEY, ZSUM, n, H);
  long aggT = (long)E * (HC / 4);
  k_agg<<<(unsigned)CDIV(aggT, 256), 256, 0, stream>>>(src, dst, LG, ZSUM, Hb, OUT, E, H, C);
  long aggS = (long)n * (HC / 4);
  k_agg_self<<<(unsigned)CDIV(aggS, 256), 256, 0, stream>>>(LGS, ZSUM, Hb, OUT, n, H, C);
}

extern "C" void kernel_launch(void* const* d_in, const int* in_sizes, int n_in,
                              void* d_out, int out_size, void* d_ws, size_t ws_size,
                              hipStream_t stream) {
  const float* x      = (const float*)d_in[0];
  const int*   ei     = (const int*)d_in[1];
  const float* ea     = (const float*)d_in[2];
  const float* fa_w   = (const float*)d_in[3];
  const float* fa_b   = (const float*)d_in[4];
  const float* W1     = (const float*)d_in[5];
  const float* We1    = (const float*)d_in[6];
  const float* as1    = (const float*)d_in[7];
  const float* ad1    = (const float*)d_in[8];
  const float* ae1    = (const float*)d_in[9];
  const float* b1     = (const float*)d_in[10];
  const float* W2     = (const float*)d_in[11];
  const float* We2    = (const float*)d_in[12];
  const float* as2    = (const float*)d_in[13];
  const float* ad2    = (const float*)d_in[14];
  const float* ae2    = (const float*)d_in[15];
  const float* b2     = (const float*)d_in[16];
  const float* W3     = (const float*)d_in[17];
  const float* We3    = (const float*)d_in[18];
  const float* as3    = (const float*)d_in[19];
  const float* ad3    = (const float*)d_in[20];
  const float* ae3    = (const float*)d_in[21];
  const float* b3     = (const float*)d_in[22];
  const float* r1_w   = (const float*)d_in[23];
  const float* r1_b   = (const float*)d_in[24];
  const float* r2_w   = (const float*)d_in[25];
  const float* r2_b   = (const float*)d_in[26];
  const float* ffn_w1 = (const float*)d_in[27];
  const float* ffn_b1 = (const float*)d_in[28];
  const float* ffn_w2 = (const float*)d_in[29];
  const float* ffn_b2 = (const float*)d_in[30];

  const int* src = ei;
  const int* dst = ei + N_EDGES;
  const int n = N_NODES, E = N_EDGES;

  // Workspace layout (fp32 elements)
  float* ws = (float*)d_ws;
  size_t o = 0;
  float* XG    = ws + o; o += (size_t)n * 128;   // gated x
  float* XR1   = ws + o; o += (size_t)n * 256;   // residual (r1, then reused for r2)
  float* Hb    = ws + o; o += (size_t)n * 256;   // per-layer h
  float* Z1    = ws + o; o += (size_t)n * 256;
  float* Z2    = ws + o; o += (size_t)n * 256;
  float* OUT   = ws + o; o += (size_t)n * 256;   // aggregation target
  float* ASRC  = ws + o; o += (size_t)n * 4;
  float* ADST  = ws + o; o += (size_t)n * 4;
  unsigned* MKEY = (unsigned*)(ws + o); o += (size_t)n * 4;
  float* ZSUM  = ws + o; o += (size_t)n * 4;
  float* LG    = ws + o; o += (size_t)E * 4;
  float* LGS   = ws + o; o += (size_t)n * 4;
  float* DEG   = ws + o; o += n;
  float* EAS   = ws + o; o += n;
  float* LOOPA = ws + o; o += n;
  float* CH    = ws + o; o += 16;
  float* F1    = ws + o; o += (size_t)n * 64;
  float* ZSELF = ws + o; o += (size_t)n * 64;
  // packed weights (dword regions)
  unsigned* Pfa  = (unsigned*)(ws + o); o += packed_dwords(128, 128);
  unsigned* Pr1  = (unsigned*)(ws + o); o += packed_dwords(128, 256);
  unsigned* PW1  = (unsigned*)(ws + o); o += packed_dwords(128, 256);
  unsigned* PW2  = (unsigned*)(ws + o); o += packed_dwords(256, 256);
  unsigned* Pr2  = (unsigned*)(ws + o); o += packed_dwords(256, 256);
  unsigned* PW3  = (unsigned*)(ws + o); o += packed_dwords(256, 64);
  unsigned* Pf1  = (unsigned*)(ws + o); o += packed_dwords(128, 64);
  unsigned* Pf2  = (unsigned*)(ws + o); o += packed_dwords(64, 64);

  // --- pre-pack all weights into WMMA tile layout ---
  pack_weight(fa_w,   Pfa, 128, 128, stream);
  pack_weight(r1_w,   Pr1, 128, 256, stream);
  pack_weight(W1,     PW1, 128, 256, stream);
  pack_weight(W2,     PW2, 256, 256, stream);
  pack_weight(r2_w,   Pr2, 256, 256, stream);
  pack_weight(W3,     PW3, 256, 64,  stream);
  pack_weight(ffn_w1, Pf1, 128, 64,  stream);
  pack_weight(ffn_w2, Pf2, 64,  64,  stream);

  // --- self-loop edge attr (mean of incoming) ---
  (void)hipMemsetAsync(DEG, 0, n * sizeof(float), stream);
  (void)hipMemsetAsync(EAS, 0, n * sizeof(float), stream);
  k_deg<<<CDIV(E, 256), 256, 0, stream>>>(src, dst, ea, DEG, EAS, E);
  k_loopattr<<<CDIV(n, 256), 256, 0, stream>>>(DEG, EAS, LOOPA, n);

  // --- feature gate: XG = x * sigmoid(x @ fa_w + fa_b) ---
  gemm_bf16_wmma<<<dim3(CDIV(n, 128), 2), 256, 0, stream>>>(
      x, Pfa, fa_b, x, XG, n, 128, 128, 1);

  // --- residual 1: XR1 = XG @ r1_w + r1_b ---
  gemm_bf16_wmma<<<dim3(CDIV(n, 128), 4), 256, 0, stream>>>(
      XG, Pr1, r1_b, nullptr, XR1, n, 256, 128, 0);

  // --- FFN branch: ZSELF = relu(XG @ ffn_w1 + ffn_b1) @ ffn_w2 + ffn_b2 ---
  gemm_bf16_wmma<<<dim3(CDIV(n, 128), 1), 256, 0, stream>>>(
      XG, Pf1, ffn_b1, nullptr, F1, n, 64, 128, 2);
  gemm_bf16_wmma<<<dim3(CDIV(n, 128), 1), 256, 0, stream>>>(
      F1, Pf2, ffn_b2, nullptr, ZSELF, n, 64, 64, 0);

  // --- layer 1: H=4, C=64, K=128 ---
  gat_layer(XG, 128, PW1, We1, as1, ad1, ae1, 4, 64, src, dst, ea, LOOPA,
            Hb, ASRC, ADST, MKEY, ZSUM, LG, LGS, CH, OUT, stream);
  k_combine<<<CDIV(n * 256, 256), 256, 0, stream>>>(OUT, b1, XR1, Z1, n, 256);

  // --- residual 2 (reuse XR1): XR1 = Z1 @ r2_w + r2_b ---
  gemm_bf16_wmma<<<dim3(CDIV(n, 128), 4), 256, 0, stream>>>(
      Z1, Pr2, r2_b, nullptr, XR1, n, 256, 256, 0);

  // --- layer 2: H=4, C=64, K=256 ---
  gat_layer(Z1, 256, PW2, We2, as2, ad2, ae2, 4, 64, src, dst, ea, LOOPA,
            Hb, ASRC, ADST, MKEY, ZSUM, LG, LGS, CH, OUT, stream);
  k_combine<<<CDIV(n * 256, 256), 256, 0, stream>>>(OUT, b2, XR1, Z2, n, 256);

  // --- layer 3: H=1, C=64, K=256 ---
  gat_layer(Z2, 256, PW3, We3, as3, ad3, ae3, 1, 64, src, dst, ea, LOOPA,
            Hb, ASRC, ADST, MKEY, ZSUM, LG, LGS, CH, OUT, stream);
  k_combine<<<CDIV(n * 64, 256), 256, 0, stream>>>(OUT, b3, ZSELF, (float*)d_out, n, 64);
}
